// PNAConv_GNNB_3092376453272
// MI455X (gfx1250) — compile-verified, matrix operand order checked by reading, not run on hardware
//
#include <hip/hip_runtime.h>
#include <hip/hip_bf16.h>

// ---------------- problem constants ----------------
#define NN 50000
#define EE 800000
#define FF 128
// E = 16*50000 edge tiles, N = 16*3125 node tiles (exact, no tails)
#define ETILES 50000
#define NTILES 3125

typedef __attribute__((ext_vector_type(16))) __bf16 v16bf;
typedef __attribute__((ext_vector_type(8)))  float  v8f;
typedef __attribute__((ext_vector_type(4))) unsigned int u32x4;
typedef __attribute__((ext_vector_type(8))) int i32x8;
typedef __attribute__((ext_vector_type(4))) int i32x4;

// 16-byte int vector + addrspace-qualified pointers for async LDS builtins
typedef int v4i_ __attribute__((vector_size(16)));
typedef __attribute__((address_space(1))) v4i_ gv4i;   // global
typedef __attribute__((address_space(3))) v4i_ lv4i;   // LDS

// CDNA5 async / tensor data-mover availability probes
#if defined(__has_builtin)
#  if __has_builtin(__builtin_amdgcn_tensor_load_to_lds) && __has_builtin(__builtin_amdgcn_s_wait_tensorcnt)
#    define USE_TDM 1
#  endif
#  if __has_builtin(__builtin_amdgcn_global_load_async_to_lds_b128)
#    define USE_ASYNC_LDS 1
#  endif
#endif

union AB { uint4 q[2]; v16bf v; };

__device__ __forceinline__ v8f wmma_bf(v16bf a, v16bf b, v8f c) {
    return __builtin_amdgcn_wmma_f32_16x16x32_bf16(false, a, false, b,
                                                   (short)0, c, false, false);
}

// round-to-nearest-even f32 -> bf16 (stored as ushort)
__device__ __forceinline__ unsigned short f2bf(float f) {
    unsigned u = __float_as_uint(f);
    u = (u + 0x7FFFu + ((u >> 16) & 1u)) >> 16;
    return (unsigned short)u;
}

// order-preserving float <-> uint key (for atomic max/min on floats)
__device__ __forceinline__ unsigned fkey(float f) {
    unsigned u = __float_as_uint(f);
    return ((int)u < 0) ? ~u : (u | 0x80000000u);
}
__device__ __forceinline__ float unfkey(unsigned k) {
    unsigned u = (k & 0x80000000u) ? (k & 0x7FFFFFFFu) : ~k;
    return __uint_as_float(u);
}

// ---------------- init / convert kernels ----------------
__global__ void k_zero_f32(float* p, int n) {
    int i = blockIdx.x * blockDim.x + threadIdx.x;
    if (i < n) p[i] = 0.0f;
}
__global__ void k_fill_u32(unsigned* p, unsigned v, int n) {
    int i = blockIdx.x * blockDim.x + threadIdx.x;
    if (i < n) p[i] = v;
}
__global__ void k_cvt_bf16(const float* __restrict__ in, unsigned short* __restrict__ out, int n) {
    int i = blockIdx.x * blockDim.x + threadIdx.x;
    if (i < n) out[i] = f2bf(in[i]);
}

// ---------------- K1: edge message GEMM + scatter aggregation ----------------
// One wave = 16 edges. m[16x128] = [x[dst]||x[src]][16x256] @ W_pre^T, bf16 WMMA,
// then atomic scatter into per-node sum / sumsq / maxkey / minkey / deg.
__global__ __launch_bounds__(128) void k_edge(
    const unsigned short* __restrict__ x_bf,     // [N][128] bf16
    const int*            __restrict__ eidx,     // [2][E]
    const unsigned short* __restrict__ wpre_bf,  // [128][256] bf16 row-major
    const float*          __restrict__ b_pre,    // [128]
    float* __restrict__ sum, float* __restrict__ sumsq, float* __restrict__ deg,
    unsigned* __restrict__ maxk, unsigned* __restrict__ mink,
    int numTiles)
{
    __shared__ unsigned short sW[128 * 256];     // 64 KB: W_pre in bf16
#ifdef USE_TDM
    // Tensor Data Mover: one DMA descriptor moves the whole 64 KB tile.
    // 2D tensor: 256 elements/row (2B each) x 128 rows, tile == tensor.
    if (threadIdx.x == 0) {
        typedef __attribute__((address_space(3))) unsigned short lds_us;
        unsigned ldsOff = (unsigned)(unsigned long long)(lds_us*)sW;
        unsigned long long ga = (unsigned long long)(const void*)wpre_bf;
        u32x4 g0 = { 1u,                                   // count=1, user desc
                     ldsOff,                               // lds_addr
                     (unsigned)(ga & 0xFFFFFFFFull),       // global_addr[31:0]
                     (unsigned)((ga >> 32) & 0x01FFFFFFull) | 0x80000000u }; // addr[56:32] | type=2
        i32x8 g1 = { (int)0x00010000u,      // wg_mask=0 | data_size=1 (2B) | no flags
                     (int)(256u << 16),     // tensor_dim0 low16 (=256)
                     (int)(128u << 16),     // tensor_dim0 hi=0 | tensor_dim1 low16 (=128)
                     (int)(256u << 16),     // tensor_dim1 hi=0 | tile_dim0=256
                     (int)128,              // tile_dim1=128 | tile_dim2=0
                     (int)256,              // tensor_dim0_stride low32 = 256
                     (int)0x80000000u,      // stride0 hi=0 | tensor_dim1_stride low16 (=32768)
                     0 };                   // tensor_dim1_stride hi = 0
        i32x4 gz4 = {0, 0, 0, 0};
#if __clang_major__ >= 23
        i32x8 gz8 = {0, 0, 0, 0, 0, 0, 0, 0};
        __builtin_amdgcn_tensor_load_to_lds(g0, g1, gz4, gz4, gz8, 0);
#else
        __builtin_amdgcn_tensor_load_to_lds(g0, g1, gz4, gz4, 0);
#endif
        __builtin_amdgcn_s_wait_tensorcnt(0);
    }
#else
    {
        const uint4* gw = (const uint4*)wpre_bf;
        uint4* lw = (uint4*)sW;
        for (int i = threadIdx.x; i < (128 * 256) / 8; i += blockDim.x) lw[i] = gw[i];
    }
#endif
    __syncthreads();

    const int wavesPerBlk = blockDim.x >> 5;
    const int wid    = blockIdx.x * wavesPerBlk + (threadIdx.x >> 5);
    const int stride = gridDim.x * wavesPerBlk;
    const int lane = threadIdx.x & 31;
    const int nl = lane & 15;      // row (A) / col (B,C)
    const int kh = lane >> 4;      // K-half selector

    const v8f vz = {0.f,0.f,0.f,0.f,0.f,0.f,0.f,0.f};

    for (int tile = wid; tile < numTiles; tile += stride) {
        const int e   = tile * 16 + nl;
        const int sId = eidx[e];        // src  (x_j)
        const int dId = eidx[EE + e];   // dst  (x_i)
        const unsigned short* rowD = x_bf + (size_t)dId * FF;
        const unsigned short* rowS = x_bf + (size_t)sId * FF;

        int nodes[8];
#pragma unroll
        for (int v = 0; v < 8; v++) nodes[v] = eidx[EE + tile * 16 + kh * 8 + v];

        v8f acc[8];
#pragma unroll
        for (int nt = 0; nt < 8; nt++) acc[nt] = vz;

#pragma unroll
        for (int kk = 0; kk < 8; kk++) {            // K = 256 in steps of 32
            const int c0 = kk * 32 + kh * 8;
            const int c1 = c0 + 16;
            AB a;
            a.q[0] = *(const uint4*)((c0 < FF) ? (rowD + c0) : (rowS + (c0 - FF)));
            a.q[1] = *(const uint4*)((c1 < FF) ? (rowD + c1) : (rowS + (c1 - FF)));
#pragma unroll
            for (int nt = 0; nt < 8; nt++) {        // N = 128 in tiles of 16
                const unsigned short* wrow = &sW[(nt * 16 + nl) * 256];
                AB b;
                b.q[0] = *(const uint4*)(wrow + c0);
                b.q[1] = *(const uint4*)(wrow + c1);
                acc[nt] = wmma_bf(a.v, b.v, acc[nt]);
            }
        }

        // scatter: C layout -> VGPR v holds row m = kh*8+v, col n = nt*16+nl
#pragma unroll
        for (int nt = 0; nt < 8; nt++) {
            const int n = nt * 16 + nl;
            const float bp = b_pre[n];
#pragma unroll
            for (int v = 0; v < 8; v++) {
                const float val = acc[nt][v] + bp;
                const size_t o = (size_t)nodes[v] * FF + n;
                __hip_atomic_fetch_add(&sum[o],   val,       __ATOMIC_RELAXED, __HIP_MEMORY_SCOPE_AGENT);
                __hip_atomic_fetch_add(&sumsq[o], val * val, __ATOMIC_RELAXED, __HIP_MEMORY_SCOPE_AGENT);
                const unsigned k = fkey(val);
                __hip_atomic_fetch_max(&maxk[o], k, __ATOMIC_RELAXED, __HIP_MEMORY_SCOPE_AGENT);
                __hip_atomic_fetch_min(&mink[o], k, __ATOMIC_RELAXED, __HIP_MEMORY_SCOPE_AGENT);
            }
        }
        if (lane < 16)
            __hip_atomic_fetch_add(&deg[dId], 1.0f, __ATOMIC_RELAXED, __HIP_MEMORY_SCOPE_AGENT);
    }
}

// ---------------- K2: per-node finalize: agg[4F] (bf16), amp/att ----------------
__global__ void k_final(
    const float* __restrict__ sum, const float* __restrict__ sumsq,
    const float* __restrict__ deg,
    const unsigned* __restrict__ maxk, const unsigned* __restrict__ mink,
    unsigned short* __restrict__ agg_bf,  // [N][512]: max|min|mean|std
    float* __restrict__ amp, float* __restrict__ att)
{
    int i = blockIdx.x * blockDim.x + threadIdx.x;
    if (i >= NN * FF) return;
    int node = i >> 7, f = i & 127;
    float d  = deg[node];
    float dc = fmaxf(d, 1.0f);
    bool  he = d > 0.0f;
    float s1 = sum[i], s2 = sumsq[i];
    float mean = s1 / dc;
    float var  = s2 / dc - mean * mean;
    float sd   = sqrtf(fmaxf(var, 0.0f) + 1e-5f);
    float mx = he ? unfkey(maxk[i]) : 0.0f;
    float mn = he ? unfkey(mink[i]) : 0.0f;
    size_t b = (size_t)node * 512;
    agg_bf[b + f]       = f2bf(mx);
    agg_bf[b + 128 + f] = f2bf(mn);
    agg_bf[b + 256 + f] = f2bf(mean);
    agg_bf[b + 384 + f] = f2bf(sd);
    if (f == 0) {
        float ld = logf(dc + 1.0f);     // DELTA = 1
        amp[node] = ld;
        att[node] = 1.0f / ld;
    }
}

// ---------------- K3: out1 = x@Wx^T + agg@W1^T + amp*(agg@W2^T) + att*(agg@W3^T) + b_post
// W_post row-major [128][1664]; col blocks: x:0..127, W1:128..639, W2:640..1151, W3:1152..1663
__global__ __launch_bounds__(128) void k_post(
    const unsigned short* __restrict__ x_bf,
    const unsigned short* __restrict__ agg_bf,
    const unsigned short* __restrict__ wpost_bf,
    const float* __restrict__ b_post,
    const float* __restrict__ amp, const float* __restrict__ att,
    unsigned short* __restrict__ out1_bf,  // [N][128] bf16
    int numTiles)
{
    const int wid = blockIdx.x * (blockDim.x >> 5) + (threadIdx.x >> 5);
    if (wid >= numTiles) return;
    const int lane = threadIdx.x & 31;
    const int nl = lane & 15, kh = lane >> 4;
    const int base = wid * 16;

    const unsigned short* xrow = x_bf   + (size_t)(base + nl) * FF;
    const unsigned short* arow = agg_bf + (size_t)(base + nl) * 512;
    __builtin_prefetch(arow, 0, 1);   // global_prefetch

    float ampv[8], attv[8];
#pragma unroll
    for (int v = 0; v < 8; v++) {
        int r = base + kh * 8 + v;
        ampv[v] = amp[r];
        attv[v] = att[r];
    }

    const v8f vz = {0.f,0.f,0.f,0.f,0.f,0.f,0.f,0.f};

    for (int nt = 0; nt < 8; nt++) {
        const unsigned short* wrow = wpost_bf + (size_t)(nt * 16 + nl) * 1664;
        v8f accU = vz, acc2 = vz, acc3 = vz;

        // x @ Wx^T  (K = 128)
#pragma unroll
        for (int kk = 0; kk < 4; kk++) {
            const int c0 = kk * 32 + kh * 8, c1 = c0 + 16;
            AB a, b;
            a.q[0] = *(const uint4*)(xrow + c0);  a.q[1] = *(const uint4*)(xrow + c1);
            b.q[0] = *(const uint4*)(wrow + c0);  b.q[1] = *(const uint4*)(wrow + c1);
            accU = wmma_bf(a.v, b.v, accU);
        }
        // agg @ {W1,W2,W3}^T  (K = 512)
#pragma unroll
        for (int kk = 0; kk < 16; kk++) {
            const int c0 = kk * 32 + kh * 8, c1 = c0 + 16;
            AB a, b1, b2, b3;
            a.q[0]  = *(const uint4*)(arow + c0);          a.q[1]  = *(const uint4*)(arow + c1);
            b1.q[0] = *(const uint4*)(wrow + 128  + c0);   b1.q[1] = *(const uint4*)(wrow + 128  + c1);
            b2.q[0] = *(const uint4*)(wrow + 640  + c0);   b2.q[1] = *(const uint4*)(wrow + 640  + c1);
            b3.q[0] = *(const uint4*)(wrow + 1152 + c0);   b3.q[1] = *(const uint4*)(wrow + 1152 + c1);
            accU = wmma_bf(a.v, b1.v, accU);
            acc2 = wmma_bf(a.v, b2.v, acc2);
            acc3 = wmma_bf(a.v, b3.v, acc3);
        }

        const int n = nt * 16 + nl;
        const float bp = b_post[n];
#pragma unroll
        for (int v = 0; v < 8; v++) {
            float val = accU[v] + ampv[v] * acc2[v] + attv[v] * acc3[v] + bp;
            int r = base + kh * 8 + v;
            out1_bf[(size_t)r * FF + n] = f2bf(val);
        }
    }
}

// ---------------- K4: out = out1 @ W_lin^T + b_lin (f32 output) ----------------
__global__ __launch_bounds__(128) void k_lin(
    const unsigned short* __restrict__ out1_bf,
    const unsigned short* __restrict__ wlin_bf,   // [128][128] bf16 row-major
    const float* __restrict__ b_lin,
    float* __restrict__ out, int numTiles)
{
    __shared__ unsigned short sWl[128 * 128];     // 32 KB: W_lin in bf16
#ifdef USE_ASYNC_LDS
    {
        for (int i = threadIdx.x; i < (128 * 128) / 8; i += blockDim.x) {
            __builtin_amdgcn_global_load_async_to_lds_b128(
                (gv4i*)(wlin_bf + i * 8),   // global src (AS1, non-const v4i*)
                (lv4i*)(sWl + i * 8),       // LDS dst
                0, 0);
        }
        asm volatile("s_wait_asynccnt 0x0" ::: "memory");
    }
#else
    {
        const uint4* gw = (const uint4*)wlin_bf;
        uint4* lw = (uint4*)sWl;
        for (int i = threadIdx.x; i < (128 * 128) / 8; i += blockDim.x) lw[i] = gw[i];
    }
#endif
    __syncthreads();

    const int wid = blockIdx.x * (blockDim.x >> 5) + (threadIdx.x >> 5);
    if (wid >= numTiles) return;
    const int lane = threadIdx.x & 31;
    const int nl = lane & 15, kh = lane >> 4;
    const int base = wid * 16;

    const unsigned short* irow = out1_bf + (size_t)(base + nl) * FF;
    const v8f vz = {0.f,0.f,0.f,0.f,0.f,0.f,0.f,0.f};

    for (int nt = 0; nt < 8; nt++) {
        const unsigned short* wrow = &sWl[(nt * 16 + nl) * FF];
        v8f acc = vz;
#pragma unroll
        for (int kk = 0; kk < 4; kk++) {
            const int c0 = kk * 32 + kh * 8, c1 = c0 + 16;
            AB a, b;
            a.q[0] = *(const uint4*)(irow + c0);  a.q[1] = *(const uint4*)(irow + c1);
            b.q[0] = *(const uint4*)(wrow + c0);  b.q[1] = *(const uint4*)(wrow + c1);
            acc = wmma_bf(a.v, b.v, acc);
        }
        const int n = nt * 16 + nl;
        const float bl = b_lin[n];
#pragma unroll
        for (int v = 0; v < 8; v++) {
            int r = base + kh * 8 + v;
            out[(size_t)r * FF + n] = acc[v] + bl;
        }
    }
}

// ---------------- host launcher ----------------
extern "C" void kernel_launch(void* const* d_in, const int* in_sizes, int n_in,
                              void* d_out, int out_size, void* d_ws, size_t ws_size,
                              hipStream_t stream) {
    const float* x      = (const float*)d_in[0];
    const int*   eidx   = (const int*)  d_in[1];
    const float* W_pre  = (const float*)d_in[2];
    const float* b_pre  = (const float*)d_in[3];
    const float* W_post = (const float*)d_in[4];
    const float* b_post = (const float*)d_in[5];
    const float* W_lin  = (const float*)d_in[6];
    const float* b_lin  = (const float*)d_in[7];
    float* out = (float*)d_out;

    const size_t NF = (size_t)NN * FF;           // 6,400,000
    const size_t NPAD = 50048;                   // N padded
    char* w = (char*)d_ws;
    size_t o = 0;
    float*          sum    = (float*)(w + o);          o += 4 * NF;
    float*          sumsq  = (float*)(w + o);          o += 4 * NF;
    float*          deg    = (float*)(w + o);          o += 4 * NPAD;
    unsigned*       maxk   = (unsigned*)(w + o);       o += 4 * NF;
    unsigned*       mink   = (unsigned*)(w + o);       o += 4 * NF;
    unsigned short* x_bf   = (unsigned short*)(w + o); o += 2 * NF;
    unsigned short* agg_bf = (unsigned short*)(w + o); o += 2 * 4 * NF;
    float*          amp    = (float*)(w + o);          o += 4 * NPAD;
    float*          att    = (float*)(w + o);          o += 4 * NPAD;
    unsigned short* wpre_bf  = (unsigned short*)(w + o); o += 2 * 128 * 256;
    unsigned short* wpost_bf = (unsigned short*)(w + o); o += 2 * 128 * 1664;
    unsigned short* wlin_bf  = (unsigned short*)(w + o); o += 2 * 128 * 128;
    unsigned short* out1_bf  = (unsigned short*)(w + o); o += 2 * NF;

    const int T = 256;
    // zero: sum | sumsq | deg | maxk are contiguous (maxkey init 0 == key(-NaN), safe floor)
    const int zc = (int)(3 * NF + NPAD);
    k_zero_f32<<<(zc + T - 1) / T, T, 0, stream>>>(sum, zc);
    k_fill_u32<<<((int)NF + T - 1) / T, T, 0, stream>>>(mink, 0xFFFFFFFFu, (int)NF);

    k_cvt_bf16<<<((int)NF + T - 1) / T, T, 0, stream>>>(x, x_bf, (int)NF);
    k_cvt_bf16<<<(128 * 256  + T - 1) / T, T, 0, stream>>>(W_pre,  wpre_bf,  128 * 256);
    k_cvt_bf16<<<(128 * 1664 + T - 1) / T, T, 0, stream>>>(W_post, wpost_bf, 128 * 1664);
    k_cvt_bf16<<<(128 * 128  + T - 1) / T, T, 0, stream>>>(W_lin,  wlin_bf,  128 * 128);

    k_edge<<<1024, 128, 0, stream>>>(x_bf, eidx, wpre_bf, b_pre,
                                     sum, sumsq, deg, maxk, mink, ETILES);

    k_final<<<((int)NF + T - 1) / T, T, 0, stream>>>(sum, sumsq, deg, maxk, mink,
                                                     agg_bf, amp, att);

    k_post<<<(NTILES + 3) / 4, 128, 0, stream>>>(x_bf, agg_bf, wpost_bf, b_post,
                                                 amp, att, out1_bf, NTILES);

    k_lin<<<(NTILES + 3) / 4, 128, 0, stream>>>(out1_bf, wlin_bf, b_lin, out, NTILES);
}
